// AttentionLayer_12189117186195
// MI455X (gfx1250) — matrix operand order repaired and checked by reading, not compiled, not run on hardware
//
#include <hip/hip_runtime.h>

// AttentionLayer for MI455X (gfx1250, wave32, WMMA, TDM).
// B=4 S=12 N=1024 D=256 H=4 G=2 HD=64 C=128.
// Pipeline: grouped proj + shuffle (f16) -> flash attention (WMMA f16, f32 acc,
// TDM double-buffered K/V tiles) -> grouped out-proj (f32 out).
// Workspace: 4 f16 tensors of 12,582,912 halves each (~100.7 MB total).

typedef __attribute__((ext_vector_type(16))) _Float16     v16h;
typedef __attribute__((ext_vector_type(8)))  float        v8f;
typedef __attribute__((ext_vector_type(4)))  unsigned int u32x4;
typedef __attribute__((ext_vector_type(8)))  int          i32x8;
typedef __attribute__((ext_vector_type(4)))  int          i32x4;
typedef __attribute__((ext_vector_type(4)))  float        f32x4;
typedef __attribute__((ext_vector_type(4)))  _Float16     h16x4;

union H16 { v16h v; u32x4 q[2]; };

#if defined(__has_builtin)
#if __has_builtin(__builtin_amdgcn_tensor_load_to_lds) && \
    __has_builtin(__builtin_amdgcn_s_wait_tensorcnt)
#define USE_TDM 1
#endif
#endif

__device__ __forceinline__ v8f wmma16(v16h a, v16h b, v8f c) {
  return __builtin_amdgcn_wmma_f32_16x16x32_f16(false, a, false, b, (short)0, c,
                                                false, false);
}

// A fragment (16x32 f16): row = lane&15; K per half-wave: sub=0 -> {0..7,16..23},
// sub=1 -> {8..15,24..31}   (ISA 7.12.2 16-bit A table)
__device__ __forceinline__ v16h fragA(const _Float16* tile, int stride, int kbase,
                                      int lane) {
  const _Float16* p = tile + (lane & 15) * stride + kbase + ((lane >> 4) & 1) * 8;
  H16 u;
  u.q[0] = *(const u32x4*)(p);
  u.q[1] = *(const u32x4*)(p + 16);
  return u.v;
}

// B fragment (32x16 f16): col = lane&15; K per half-wave: sub=0 -> {0..7,16..23}, sub=1 -> {8..15,24..31}
__device__ __forceinline__ v16h fragB(const _Float16* tile, int stride, int kbase,
                                      int lane) {
  const _Float16* p = tile + (lane & 15) * stride + kbase + ((lane >> 4) & 1) * 8;
  H16 u;
  u.q[0] = *(const u32x4*)(p);
  u.q[1] = *(const u32x4*)(p + 16);
  return u.v;
}

#ifdef USE_TDM
// Issue a TDM 2D tile load: tile_d1 rows of tile_d0 f16 elements, row stride
// stride0 elements, into contiguous LDS at lds_off. D# per ISA cdna5 8.3/8.4.
// This toolchain exposes the 6-arg builtin:
//   (uint32x4 g0, int32x8 g1, int32x4 g2, int32x4 g3, int32x8 g4, i32 cpol)
__device__ __forceinline__ void tdm_load_2d(unsigned lds_off, const void* gsrc,
                                            unsigned tile_d0, unsigned tile_d1,
                                            unsigned long long stride0) {
  unsigned long long ga = (unsigned long long)(size_t)gsrc;
  u32x4 g0;
  g0.x = 1u;                                               // count=1 valid user D#
  g0.y = lds_off;                                          // lds_addr (bytes)
  g0.z = (unsigned)(ga & 0xffffffffu);                     // global_addr[31:0]
  g0.w = (unsigned)((ga >> 32) & 0x01ffffffu) | (2u << 30);  // addr[56:32], type=2
  i32x8 g1;
  g1[0] = (int)(1u << 16);                                 // data_size=1 -> 2 bytes
  g1[1] = (int)((tile_d0 & 0xffffu) << 16);                // tensor_dim0[15:0]
  g1[2] = (int)(((tile_d0 >> 16) & 0xffffu) |              // tensor_dim0[31:16]
                ((tile_d1 & 0xffffu) << 16));              // tensor_dim1[15:0]
  g1[3] = (int)(((tile_d1 >> 16) & 0xffffu) |              // tensor_dim1[31:16]
                ((tile_d0 & 0xffffu) << 16));              // tile_dim0
  g1[4] = (int)(tile_d1 & 0xffffu);                        // tile_dim1 (tile_dim2=0)
  g1[5] = (int)(unsigned)(stride0 & 0xffffffffu);          // tensor_dim0_stride lo
  g1[6] = (int)(unsigned)((stride0 >> 32) & 0xffffu);      // stride hi (dim1_stride=0)
  g1[7] = 0;
  i32x4 gz4 = {0, 0, 0, 0};
  i32x8 gz8 = {0, 0, 0, 0, 0, 0, 0, 0};
  __builtin_amdgcn_tensor_load_to_lds(g0, g1, gz4, gz4, gz8, 0);
}
__device__ __forceinline__ unsigned lds_off(const void* p) {
  return (unsigned)(size_t)p;  // generic LDS pointer: low 32 bits = LDS byte offset
}
#endif

// ---------------- Kernel 1: grouped FC + channel shuffle + head split ----------
// rows r in [0,98304): r = ((b*12+s)*1024 + n)*2 + g ; y[r][oc]=sum_k x[r][k]W[oc][k]+b
// dest: t=g*12+s; s2=t>>1; n2=(t&1)*512+(n>>1); d2=(n&1)*128+oc; h=d2>>6; hd=d2&63
__global__ __launch_bounds__(256) void proj_shuffle_kernel(
    const float* __restrict__ X, const float* __restrict__ W,
    const float* __restrict__ bias, _Float16* __restrict__ out, const int vtrans) {
  __shared__ __align__(16) _Float16 ldsA[128 * 128];
  __shared__ __align__(16) _Float16 ldsW[128 * 128];
  __shared__ float ldsBias[128];

  const int tid = threadIdx.x;
  const int r0  = blockIdx.x * 128;

  const f32x4* Ab = (const f32x4*)(X + (size_t)r0 * 128);
  const f32x4* Wb = (const f32x4*)W;
#pragma unroll
  for (int i = 0; i < 16; ++i) {
    int   qd = i * 256 + tid;
    f32x4 fa = Ab[qd];
    f32x4 fw = Wb[qd];
    h16x4 ha = {(_Float16)fa.x, (_Float16)fa.y, (_Float16)fa.z, (_Float16)fa.w};
    h16x4 hw = {(_Float16)fw.x, (_Float16)fw.y, (_Float16)fw.z, (_Float16)fw.w};
    *(h16x4*)(ldsA + qd * 4) = ha;
    *(h16x4*)(ldsW + qd * 4) = hw;
  }
  if (tid < 128) ldsBias[tid] = bias[tid];
  __syncthreads();

  const int wave = tid >> 5, lane = tid & 31;
  const int sub = (lane >> 4) & 1, l15 = lane & 15;

  v16h a[4];
#pragma unroll
  for (int ks = 0; ks < 4; ++ks)
    a[ks] = fragA(ldsA + wave * 16 * 128, 128, ks * 32, lane);

  // hoisted shuffle decode: idx = basej[j] + off(ct)
  unsigned basej[8];
#pragma unroll
  for (int j = 0; j < 8; ++j) {
    int grow = r0 + wave * 16 + j + 8 * sub;
    int g  = grow & 1;
    int rn = grow >> 1;
    int n  = rn & 1023;
    int bs = rn >> 10;
    int s  = bs % 12;
    int bb = bs / 12;
    int t  = g * 12 + s;
    int P  = bb * 12 + (t >> 1);                       // b*12 + s2
    int n2 = ((t & 1) << 9) | (n >> 1);
    int e  = n & 1;
    basej[j] = (unsigned)(P * 262144 + e * 131072 + (vtrans ? n2 : n2 * 64));
  }

#pragma unroll
  for (int ct = 0; ct < 8; ++ct) {
    v8f c = {};
#pragma unroll
    for (int ks = 0; ks < 4; ++ks)
      c = wmma16(a[ks], fragB(ldsW + ct * 16 * 128, 128, ks * 32, lane), c);

    const int      oc  = ct * 16 + l15;
    const float    bv  = ldsBias[oc];
    const unsigned off = (unsigned)((ct >> 2) * 65536 +
                                    ((ct & 3) * 16 + l15) * (vtrans ? 1024 : 1));
#pragma unroll
    for (int j = 0; j < 8; ++j) out[basej[j] + off] = (_Float16)(c[j] + bv);
  }
}

// ---------------- Kernel 2: flash attention over nodes -------------------------
// q16/k16: [p][n][hd] (p = (b*12+s2)*4+h, 65536 halves per p); v16: [p][hd][n]
__global__ __launch_bounds__(256) void attention_kernel(
    const _Float16* __restrict__ q16, const _Float16* __restrict__ k16,
    const _Float16* __restrict__ v16, _Float16* __restrict__ aout) {
  __shared__ __align__(16) _Float16 ldsQ[128 * 64];
  __shared__ __align__(16) _Float16 ldsK[2][128 * 64];
  __shared__ __align__(16) _Float16 ldsVt[2][64 * 128];
  __shared__ __align__(16) _Float16 ldsP[8 * 16 * 128];

  const int tid = threadIdx.x;
  const int p   = blockIdx.x >> 3;
  const int qt  = blockIdx.x & 7;
  const int b   = p / 48;
  const int s2  = (p >> 2) % 12;
  const int h   = p & 3;

  const _Float16* qb = q16 + (size_t)p * 65536;
  const _Float16* kb = k16 + (size_t)p * 65536;
  const _Float16* vb = v16 + (size_t)p * 65536;

  const int wave = tid >> 5, lane = tid & 31;
  const int sub = (lane >> 4) & 1, l15 = lane & 15;
  const int pOff = wave * 16 * 128;

#ifdef USE_TDM
  if (wave == 0) {  // TDM: Q tile + first two K/Vt tile pairs (TENSORcnt in-order)
    tdm_load_2d(lds_off(ldsQ), qb + qt * 8192, 8192, 1, 8192);
    tdm_load_2d(lds_off(ldsK[0]), kb, 8192, 1, 8192);
    tdm_load_2d(lds_off(ldsVt[0]), vb, 128, 64, 1024);
    tdm_load_2d(lds_off(ldsK[1]), kb + 8192, 8192, 1, 8192);
    tdm_load_2d(lds_off(ldsVt[1]), vb + 128, 128, 64, 1024);
  }
#else
  {  // Q tile 128x64
    const u32x4* src = (const u32x4*)(qb + qt * 8192);
    u32x4*       dst = (u32x4*)ldsQ;
#pragma unroll
    for (int i = 0; i < 4; ++i) dst[i * 256 + tid] = src[i * 256 + tid];
  }
#endif

  v16h  aq0 = {}, aq1 = {};
  v8f   acc[4] = {};
  float mrow[8], lrow[8];
#pragma unroll
  for (int j = 0; j < 8; ++j) { mrow[j] = -3.0e38f; lrow[j] = 0.f; }

  for (int kt = 0; kt < 8; ++kt) {
#ifdef USE_TDM
    if (wave == 0) {  // pair kt complete when <=2 ops (pair kt+1) remain
      if (kt < 7) __builtin_amdgcn_s_wait_tensorcnt(2);
      else        __builtin_amdgcn_s_wait_tensorcnt(0);
    }
    __syncthreads();
#else
    __syncthreads();
    {  // K tile 128x64
      const u32x4* src = (const u32x4*)(kb + kt * 8192);
      u32x4*       dst = (u32x4*)ldsK[kt & 1];
#pragma unroll
      for (int i = 0; i < 4; ++i) dst[i * 256 + tid] = src[i * 256 + tid];
    }
#pragma unroll
    for (int i = 0; i < 4; ++i) {  // V^T tile: 64 rows x 128 keys
      int c = i * 256 + tid;
      int row = c >> 4, ch = c & 15;
      ((u32x4*)ldsVt[kt & 1])[c] =
          *(const u32x4*)(vb + row * 1024 + kt * 128 + ch * 8);
    }
    if (kt < 7) {
      __builtin_prefetch(kb + (kt + 1) * 8192 + tid * 32, 0, 0);
      __builtin_prefetch(vb + (tid >> 2) * 1024 + (kt + 1) * 128 + (tid & 3) * 32, 0, 0);
    }
    __syncthreads();
#endif
    if (kt == 0) {
      aq0 = fragA(ldsQ + wave * 16 * 64, 64, 0, lane);
      aq1 = fragA(ldsQ + wave * 16 * 64, 64, 32, lane);
    }
    const _Float16* Kc = ldsK[kt & 1];
    const _Float16* Vc = ldsVt[kt & 1];

    v8f st[8];  // 16 q-rows x 128 keys, scaled by 1/sqrt(64)
#pragma unroll
    for (int ct = 0; ct < 8; ++ct) {
      v8f c = {};
      c = wmma16(aq0, fragB(Kc + ct * 16 * 64, 64, 0, lane), c);
      c = wmma16(aq1, fragB(Kc + ct * 16 * 64, 64, 32, lane), c);
      st[ct] = c * 0.125f;
    }

#pragma unroll
    for (int j = 0; j < 8; ++j) {  // online softmax; row = j + 8*sub
      float mx = mrow[j];
#pragma unroll
      for (int ct = 0; ct < 8; ++ct) mx = fmaxf(mx, st[ct][j]);
#pragma unroll
      for (int off = 1; off < 16; off <<= 1) mx = fmaxf(mx, __shfl_xor(mx, off, 32));
      const float corr = __expf(mrow[j] - mx);
      mrow[j] = mx;
      float     ls   = 0.f;
      _Float16* prow = ldsP + pOff + (j + 8 * sub) * 128 + l15;
#pragma unroll
      for (int ct = 0; ct < 8; ++ct) {
        float e = __expf(st[ct][j] - mx);
        ls += e;
        prow[ct * 16] = (_Float16)e;
      }
      lrow[j] = lrow[j] * corr + ls;
#pragma unroll
      for (int ot = 0; ot < 4; ++ot) acc[ot][j] *= corr;
    }

#pragma unroll
    for (int ot = 0; ot < 4; ++ot)  // acc += P (16x128) @ V (128x64)
#pragma unroll
      for (int ks = 0; ks < 4; ++ks)
        acc[ot] = wmma16(fragA(ldsP + pOff, 128, ks * 32, lane),
                         fragB(Vc + ot * 16 * 128, 128, ks * 32, lane), acc[ot]);

#ifdef USE_TDM
    __syncthreads();  // all waves done reading buf[kt&1] before TDM overwrites it
    if (wave == 0 && kt + 2 < 8) {
      tdm_load_2d(lds_off(ldsK[kt & 1]), kb + (kt + 2) * 8192, 8192, 1, 8192);
      tdm_load_2d(lds_off(ldsVt[kt & 1]), vb + (kt + 2) * 128, 128, 64, 1024);
    }
#endif
  }

#pragma unroll
  for (int j = 0; j < 8; ++j) {
    float l = lrow[j];
#pragma unroll
    for (int off = 1; off < 16; off <<= 1) l += __shfl_xor(l, off, 32);
    const float inv = 1.f / l;
#pragma unroll
    for (int ot = 0; ot < 4; ++ot) acc[ot][j] *= inv;
  }

  // store to [b][s2][n2][h*64+d] f16
  const int    n2base = qt * 128 + wave * 16 + 8 * sub;
  const size_t obase  = ((size_t)(b * 12 + s2) * 1024) * 256 + h * 64;
#pragma unroll
  for (int ot = 0; ot < 4; ++ot)
#pragma unroll
    for (int j = 0; j < 8; ++j)
      aout[obase + (size_t)(n2base + j) * 256 + ot * 16 + l15] = (_Float16)acc[ot][j];
}

// ---------------- Kernel 3: grouped output projection (fp32 out) ---------------
__global__ __launch_bounds__(256) void outproj_kernel(
    const _Float16* __restrict__ A16, const float* __restrict__ W,
    const float* __restrict__ bias, float* __restrict__ out) {
  __shared__ __align__(16) _Float16 ldsA[128 * 128];
  __shared__ __align__(16) _Float16 ldsW[128 * 128];
  __shared__ float ldsBias[128];

  const int tid = threadIdx.x;
  const int r0  = blockIdx.x * 128;
  const int wave = tid >> 5, lane = tid & 31;
  const int sub = (lane >> 4) & 1, l15 = lane & 15;

#ifdef USE_TDM
  if (wave == 0)  // contiguous 32KB f16 A tile via TDM; overlaps W conversion
    tdm_load_2d(lds_off(ldsA), A16 + (size_t)r0 * 128, 16384, 1, 16384);
#else
  {
    const u32x4* src = (const u32x4*)(A16 + (size_t)r0 * 128);
    u32x4*       dst = (u32x4*)ldsA;
#pragma unroll
    for (int i = 0; i < 8; ++i) dst[i * 256 + tid] = src[i * 256 + tid];
  }
#endif
  const f32x4* Wb = (const f32x4*)W;
#pragma unroll
  for (int i = 0; i < 16; ++i) {
    int   qd = i * 256 + tid;
    f32x4 fw = Wb[qd];
    h16x4 hw = {(_Float16)fw.x, (_Float16)fw.y, (_Float16)fw.z, (_Float16)fw.w};
    *(h16x4*)(ldsW + qd * 4) = hw;
  }
  if (tid < 128) ldsBias[tid] = bias[tid];
#ifdef USE_TDM
  if (wave == 0) __builtin_amdgcn_s_wait_tensorcnt(0);
#endif
  __syncthreads();

  v16h a[4];
#pragma unroll
  for (int ks = 0; ks < 4; ++ks)
    a[ks] = fragA(ldsA + wave * 16 * 128, 128, ks * 32, lane);

#pragma unroll
  for (int ct = 0; ct < 8; ++ct) {
    v8f c = {};
#pragma unroll
    for (int ks = 0; ks < 4; ++ks)
      c = wmma16(a[ks], fragB(ldsW + ct * 16 * 128, 128, ks * 32, lane), c);
    const int   oc = ct * 16 + l15;
    const float bv = ldsBias[oc];
#pragma unroll
    for (int j = 0; j < 8; ++j) {
      int grow = r0 + wave * 16 + j + 8 * sub;
      out[(size_t)grow * 128 + oc] = c[j] + bv;
    }
  }
}

extern "C" void kernel_launch(void* const* d_in, const int* in_sizes, int n_in,
                              void* d_out, int out_size, void* d_ws, size_t ws_size,
                              hipStream_t stream) {
  const float* q  = (const float*)d_in[0];
  const float* k  = (const float*)d_in[1];
  const float* v  = (const float*)d_in[2];
  const float* Wq = (const float*)d_in[3];
  const float* bq = (const float*)d_in[4];
  const float* Wk = (const float*)d_in[5];
  const float* bk = (const float*)d_in[6];
  const float* Wv = (const float*)d_in[7];
  const float* bv = (const float*)d_in[8];
  const float* Wo = (const float*)d_in[9];
  const float* bo = (const float*)d_in[10];
  float*       out = (float*)d_out;

  const size_t T   = (size_t)4 * 12 * 4 * 1024 * 64;  // 12,582,912 halves per tensor
  _Float16*    ws  = (_Float16*)d_ws;
  _Float16*    q16 = ws;
  _Float16*    k16 = ws + T;
  _Float16*    v16 = ws + 2 * T;
  _Float16*    a16 = ws + 3 * T;  // needs ws_size >= 8*T bytes (~100.7 MB)

  dim3 blk(256);
  proj_shuffle_kernel<<<dim3(768), blk, 0, stream>>>(q, Wq, bq, q16, 0);
  proj_shuffle_kernel<<<dim3(768), blk, 0, stream>>>(k, Wk, bk, k16, 0);
  proj_shuffle_kernel<<<dim3(768), blk, 0, stream>>>(v, Wv, bv, v16, 1);
  attention_kernel<<<dim3(1536), blk, 0, stream>>>(q16, k16, v16, a16);
  outproj_kernel<<<dim3(768), blk, 0, stream>>>(a16, Wo, bo, out);
}
